// Block3DAttention_30477087932846
// MI455X (gfx1250) — matrix operand order repaired and checked by reading, not compiled, not run on hardware
//
#include <hip/hip_runtime.h>

// ---------------------------------------------------------------------------
// Block3D attention for MI455X (gfx1250), wave32 + WMMA bf16 16x16x32.
//   B=2, L=8192, HID=2048, NH=32, HD=64, block_len=256, nblocks=64
// Pipeline:
//   1) fp32 -> bf16 converts (hs, Wq, Wk, Wv, Wo)
//   2) fused QKV GEMM (bf16 WMMA, f32 accum)
//   3) per-(block,head) attention (WMMA S, softmax, WMMA P*V)
//   4) output GEMM -> f32 d_out
// ---------------------------------------------------------------------------

typedef __bf16 bf16;
typedef __attribute__((ext_vector_type(16))) __bf16 v16bf;
typedef __attribute__((ext_vector_type(8)))  __bf16 v8bf;
typedef __attribute__((ext_vector_type(8)))  float  v8f;

#define HIDDIM 2048
#define NHEADS 32
#define HEADD  64
#define BLKLEN 256
#define NBLK   64
#define MROWS  16384   // B * L

__device__ __forceinline__ v8f wmma_bf16(v16bf a, v16bf b, v8f c) {
  // D = A(16x32) * B(32x16) + C(16x16 f32)
  return __builtin_amdgcn_wmma_f32_16x16x32_bf16(
      /*neg_a=*/false, a, /*neg_b=*/false, b,
      /*c_mod=*/(short)0, c, /*reuse_a=*/false, /*reuse_b=*/false);
}

// A-matrix 16x32 bf16 fragment: lane m = lane&15, half = lane>>4
//   elements 0..7  <- K = half*16 + 0..7
//   elements 8..15 <- K = half*16 + 8 + 0..7
__device__ __forceinline__ v16bf load_a16(const bf16* rowptr, int half) {
  v8bf lo = *(const v8bf*)(rowptr + half * 16);
  v8bf hi = *(const v8bf*)(rowptr + half * 16 + 8);
  return __builtin_shufflevector(lo, hi, 0,1,2,3,4,5,6,7,8,9,10,11,12,13,14,15);
}

// Sequence-row index for the blockify rearrange:
//   s = bnz*4096 + sz*1024 + bx*256 + sx*32 + by*8 + sy,  n = ((b*2+bnz)*4+bx)*4+by
__device__ __forceinline__ int seq_row(int n, int l) {
  int b_i = n >> 5;
  int nl  = n & 31;
  int bnz = nl >> 4;
  int bx  = (nl >> 2) & 3;
  int by  = nl & 3;
  int sz  = l >> 6;
  int sx  = (l >> 3) & 7;
  int sy  = l & 7;
  return (b_i << 13) + (bnz << 12) + (sz << 10) + (bx << 8) + (sx << 5) + (by << 3) + sy;
}

// ---------------------------------------------------------------------------
// Kernel 1: fp32 -> bf16 convert (grid-stride)
// ---------------------------------------------------------------------------
__global__ __launch_bounds__(256)
void k_f32_to_bf16(const float* __restrict__ in, bf16* __restrict__ out, int n) {
  int stride = gridDim.x * blockDim.x;
  for (int i = blockIdx.x * blockDim.x + threadIdx.x; i < n; i += stride)
    out[i] = (bf16)in[i];
}

// ---------------------------------------------------------------------------
// Kernel 2: fused QKV GEMM.  C[m,nc] = sum_k X[m,k] * W{q,k,v}[nc,k]
// Workgroup: 256 threads = 8 waves as 4(M) x 2(N); wave tile 32x32; WG 128x64.
// Grid: (16384/128, 2048/64) = (128, 32)
// ---------------------------------------------------------------------------
__global__ __launch_bounds__(256)
void k_qkv_gemm(const bf16* __restrict__ X,
                const bf16* __restrict__ Wq, const bf16* __restrict__ Wk,
                const bf16* __restrict__ Wv,
                bf16* __restrict__ Q, bf16* __restrict__ K, bf16* __restrict__ V) {
  const int lane = threadIdx.x & 31;
  const int wave = threadIdx.x >> 5;
  const int wm = wave & 3, wn = wave >> 2;
  const int M0 = blockIdx.x * 128 + wm * 32;
  const int N0 = blockIdx.y * 64  + wn * 32;
  const int half = lane >> 4, lm = lane & 15;

  v8f aq[2][2] = {}, ak[2][2] = {}, av[2][2] = {};

  for (int k0 = 0; k0 < HIDDIM; k0 += 32) {
    v16bf a[2];
#pragma unroll
    for (int ti = 0; ti < 2; ++ti) {
      const bf16* rp = X + (size_t)(M0 + ti * 16 + lm) * HIDDIM + k0;
      a[ti] = load_a16(rp, half);
    }
#pragma unroll
    for (int tj = 0; tj < 2; ++tj) {
      int nc = N0 + tj * 16 + lm;
      size_t woff = (size_t)nc * HIDDIM + k0 + half * 16;
      v16bf bq = *(const v16bf*)(Wq + woff);
      v16bf bk = *(const v16bf*)(Wk + woff);
      v16bf bv = *(const v16bf*)(Wv + woff);
#pragma unroll
      for (int ti = 0; ti < 2; ++ti) {
        aq[ti][tj] = wmma_bf16(a[ti], bq, aq[ti][tj]);
        ak[ti][tj] = wmma_bf16(a[ti], bk, ak[ti][tj]);
        av[ti][tj] = wmma_bf16(a[ti], bv, av[ti][tj]);
      }
    }
  }

  // C/D layout: lanes 0-15: (M=r, N=lane); lanes 16-31: (M=r+8, N=lane-16)
  const int moff = half * 8;
#pragma unroll
  for (int ti = 0; ti < 2; ++ti)
#pragma unroll
    for (int tj = 0; tj < 2; ++tj) {
      int col = N0 + tj * 16 + lm;
#pragma unroll
      for (int r = 0; r < 8; ++r) {
        size_t off = (size_t)(M0 + ti * 16 + moff + r) * HIDDIM + col;
        Q[off] = (bf16)aq[ti][tj][r];
        K[off] = (bf16)ak[ti][tj][r];
        V[off] = (bf16)av[ti][tj][r];
      }
    }
}

// ---------------------------------------------------------------------------
// Kernel 3: per-(block, head, 64-row panel) attention.
// Block: 128 threads = 4 waves, each wave owns 16 query rows.
// Grid: 64 blocks * 32 heads * 4 panels = 8192.
// LDS: VT (V transposed, 64x256 bf16, 32KB) + P panel (64x256 bf16, 32KB).
// ---------------------------------------------------------------------------
__global__ __launch_bounds__(128)
void k_block_attn(const bf16* __restrict__ Qb, const bf16* __restrict__ Kb,
                  const bf16* __restrict__ Vb, bf16* __restrict__ O) {
  __shared__ bf16 VT[64 * 256];   // VT[d][k] = V[k][d]
  __shared__ bf16 Pl[64 * 256];   // softmax(P) panel, row-major

  const int blk   = blockIdx.x;
  const int panel = blk & 3;
  const int h     = (blk >> 2) & 31;
  const int n     = blk >> 7;

  const int tid  = threadIdx.x;
  const int lane = tid & 31;
  const int wave = tid >> 5;
  const int half = lane >> 4, lm = lane & 15;
  const float scale = 0.125f;  // 1/sqrt(64)

  // Stage V^T into LDS (whole 256x64 block for this head).
  for (int idx = tid; idx < BLKLEN * HEADD; idx += 128) {
    int kk = idx >> 6, d = idx & 63;
    int srow = seq_row(n, kk);
    VT[d * 256 + kk] = Vb[(size_t)srow * HIDDIM + h * HEADD + d];
  }

  // ---- Pass 1: S = Q K^T for this wave's 16 rows ----
  const int m0 = panel * 64 + wave * 16;
  int qrow = seq_row(n, m0 + lm);
  const bf16* qp = Qb + (size_t)qrow * HIDDIM + h * HEADD;
  v16bf a0 = load_a16(qp, half);       // k = 0..31
  v16bf a1 = load_a16(qp + 32, half);  // k = 32..63

  v8f sacc[16] = {};
#pragma unroll
  for (int j = 0; j < 16; ++j) {
    int crow = seq_row(n, j * 16 + lm);
    const bf16* kp = Kb + (size_t)crow * HIDDIM + h * HEADD + half * 16;
    v16bf b0 = *(const v16bf*)kp;
    v16bf b1 = *(const v16bf*)(kp + 32);
    sacc[j] = wmma_bf16(a0, b0, sacc[j]);
    sacc[j] = wmma_bf16(a1, b1, sacc[j]);
  }

  // ---- Softmax over 256 cols per row (rows live at fixed r, spread over
  //      16 lanes of one half-wave; reduce over j-tiles then lanes) ----
  float mx[8], sm[8], inv[8];
#pragma unroll
  for (int r = 0; r < 8; ++r) {
    float m = -3.0e38f;
#pragma unroll
    for (int j = 0; j < 16; ++j) m = fmaxf(m, sacc[j][r]);
#pragma unroll
    for (int off = 8; off >= 1; off >>= 1)
      m = fmaxf(m, __shfl_xor(m, off, 32));
    mx[r] = m;
  }
#pragma unroll
  for (int r = 0; r < 8; ++r) {
    float s = 0.0f;
#pragma unroll
    for (int j = 0; j < 16; ++j) {
      float p = __expf(scale * (sacc[j][r] - mx[r]));
      sacc[j][r] = p;
      s += p;
    }
#pragma unroll
    for (int off = 8; off >= 1; off >>= 1)
      s += __shfl_xor(s, off, 32);
    sm[r] = s;
    inv[r] = 1.0f / s;
  }

  // Write normalized P (bf16) to LDS panel, row-major.
#pragma unroll
  for (int j = 0; j < 16; ++j) {
    int col = j * 16 + lm;
#pragma unroll
    for (int r = 0; r < 8; ++r) {
      int rowl = wave * 16 + r + half * 8;
      Pl[rowl * 256 + col] = (bf16)(sacc[j][r] * inv[r]);
    }
  }

  __syncthreads();

  // ---- Pass 2: O = P * V  (A from Pl LDS, B from VT LDS) ----
  const bf16* prow = Pl + (wave * 16 + lm) * 256;
  v8f oacc[4] = {};
  for (int kk = 0; kk < 256; kk += 32) {
    v16bf a = load_a16(prow + kk, half);
#pragma unroll
    for (int dt = 0; dt < 4; ++dt) {
      const bf16* vp = VT + (dt * 16 + lm) * 256 + kk + half * 16;
      v16bf b = *(const v16bf*)vp;
      oacc[dt] = wmma_bf16(a, b, oacc[dt]);
    }
  }

  // Store O at the reference's plain-reshape location: row = n*256 + m.
  const int moff = half * 8;
#pragma unroll
  for (int dt = 0; dt < 4; ++dt) {
    int col = h * HEADD + dt * 16 + lm;
#pragma unroll
    for (int r = 0; r < 8; ++r) {
      size_t orow = (size_t)n * BLKLEN + panel * 64 + wave * 16 + moff + r;
      O[orow * HIDDIM + col] = (bf16)oacc[dt][r];
    }
  }
}

// ---------------------------------------------------------------------------
// Kernel 4: output GEMM. out[m,nc] = sum_k O[m,k] * Wo[nc,k]  (f32 out)
// Same tiling as QKV GEMM; grid (128, 32), 256 threads.
// ---------------------------------------------------------------------------
__global__ __launch_bounds__(256)
void k_out_gemm(const bf16* __restrict__ X, const bf16* __restrict__ Wo,
                float* __restrict__ out) {
  const int lane = threadIdx.x & 31;
  const int wave = threadIdx.x >> 5;
  const int wm = wave & 3, wn = wave >> 2;
  const int M0 = blockIdx.x * 128 + wm * 32;
  const int N0 = blockIdx.y * 64  + wn * 32;
  const int half = lane >> 4, lm = lane & 15;

  v8f acc[2][2] = {};
  for (int k0 = 0; k0 < HIDDIM; k0 += 32) {
    v16bf a[2];
#pragma unroll
    for (int ti = 0; ti < 2; ++ti) {
      const bf16* rp = X + (size_t)(M0 + ti * 16 + lm) * HIDDIM + k0;
      a[ti] = load_a16(rp, half);
    }
#pragma unroll
    for (int tj = 0; tj < 2; ++tj) {
      int nc = N0 + tj * 16 + lm;
      v16bf b = *(const v16bf*)(Wo + (size_t)nc * HIDDIM + k0 + half * 16);
#pragma unroll
      for (int ti = 0; ti < 2; ++ti)
        acc[ti][tj] = wmma_bf16(a[ti], b, acc[ti][tj]);
    }
  }

  const int moff = half * 8;
#pragma unroll
  for (int ti = 0; ti < 2; ++ti)
#pragma unroll
    for (int tj = 0; tj < 2; ++tj) {
      int col = N0 + tj * 16 + lm;
#pragma unroll
      for (int r = 0; r < 8; ++r)
        out[(size_t)(M0 + ti * 16 + moff + r) * HIDDIM + col] = acc[ti][tj][r];
    }
}

// ---------------------------------------------------------------------------
// Launcher
// ---------------------------------------------------------------------------
extern "C" void kernel_launch(void* const* d_in, const int* in_sizes, int n_in,
                              void* d_out, int out_size, void* d_ws, size_t ws_size,
                              hipStream_t stream) {
  const float* hs = (const float*)d_in[0];
  const float* Wq = (const float*)d_in[1];
  const float* Wk = (const float*)d_in[2];
  const float* Wv = (const float*)d_in[3];
  const float* Wo = (const float*)d_in[4];

  const size_t ACT = (size_t)MROWS * HIDDIM;     // 33,554,432 elements
  const size_t WEL = (size_t)HIDDIM * HIDDIM;    //  4,194,304 elements

  bf16* hsb = (bf16*)d_ws;
  bf16* wqb = hsb + ACT;
  bf16* wkb = wqb + WEL;
  bf16* wvb = wkb + WEL;
  bf16* wob = wvb + WEL;
  bf16* qb  = wob + WEL;
  bf16* kb  = qb  + ACT;
  bf16* vb  = kb  + ACT;
  bf16* ob  = vb  + ACT;

  // 1) converts
  k_f32_to_bf16<<<4096, 256, 0, stream>>>(hs, hsb, (int)ACT);
  k_f32_to_bf16<<<2048, 256, 0, stream>>>(Wq, wqb, (int)WEL);
  k_f32_to_bf16<<<2048, 256, 0, stream>>>(Wk, wkb, (int)WEL);
  k_f32_to_bf16<<<2048, 256, 0, stream>>>(Wv, wvb, (int)WEL);
  k_f32_to_bf16<<<2048, 256, 0, stream>>>(Wo, wob, (int)WEL);

  // 2) fused QKV projection
  k_qkv_gemm<<<dim3(MROWS / 128, HIDDIM / 64), 256, 0, stream>>>(
      hsb, wqb, wkb, wvb, qb, kb, vb);

  // 3) block attention: 64 blocks * 32 heads * 4 row-panels
  k_block_attn<<<NBLK * NHEADS * 4, 128, 0, stream>>>(qb, kb, vb, ob);

  // 4) output projection -> f32
  k_out_gemm<<<dim3(MROWS / 128, HIDDIM / 64), 256, 0, stream>>>(
      ob, wob, (float*)d_out);
}
